// MaskedAttentionBlock_8615704395976
// MI455X (gfx1250) — matrix operand order repaired and checked by reading, not compiled, not run on hardware
//
#include <hip/hip_runtime.h>

// ---------------- problem constants ----------------
#define HIDN     256
#define HEADS    8
#define HEAD_DIM 32
#define BATCH    2
#define SEQ      2048
#define MROWS    (BATCH * SEQ)   // 4096 token rows
#define FF       1024
#define NEGINF   (-1e9f)

typedef _Float16 f16;
typedef __attribute__((ext_vector_type(16))) _Float16 v16h;
typedef __attribute__((ext_vector_type(8)))  _Float16 v8h;
typedef __attribute__((ext_vector_type(8)))  float    v8f;

union V16U { v16h v; v8h h[2]; };

// ---------------- WMMA fragment helpers (wave32, 16x16x32 f16) ----------------
// A fragment (16 rows x 32 K, f16): lane&15 = row M; lane>>4 selects K-halves:
//   lanes 0-15 hold K 0..7 and 16..23; lanes 16-31 hold K 8..15 and 24..31.
__device__ __forceinline__ v16h load_a_frag(const f16* base, int ld, int kBase, int lane) {
  int row = lane & 15;
  int ko  = kBase + ((lane >> 4) << 3);
  const f16* p = base + (size_t)row * ld + ko;
  V16U r;
  r.h[0] = *(const v8h*)(p);        // K = ko .. ko+7
  r.h[1] = *(const v8h*)(p + 16);   // K = ko+16 .. ko+23
  return r.v;
}

// B fragment (32 K x 16 cols, f16) from K-major storage W[col][k]:
//   lane&15 = column N; lanes 0-15 hold K 0..7 and 16..23; lanes 16-31 hold K 8..15 and 24..31.
__device__ __forceinline__ v16h load_b_frag(const f16* W, int ldk, int nBase, int kBase, int lane) {
  int col = nBase + (lane & 15);
  int ko  = kBase + ((lane >> 4) << 3);
  const f16* p = W + (size_t)col * ldk + ko;
  V16U r;
  r.h[0] = *(const v8h*)(p);
  r.h[1] = *(const v8h*)(p + 16);
  return r.v;
}

__device__ __forceinline__ v8f wmma16(v16h a, v16h b, v8f c) {
  // D = A(16x32 f16) * B(32x16 f16) + C(f32)
  return __builtin_amdgcn_wmma_f32_16x16x32_f16(false, a, false, b, (short)0, c, false, false);
}

// half-wave (16-lane) reductions: xor masks <= 8 stay inside a lane-half,
// which is exactly how C/D fragment rows are striped on gfx1250 wave32.
__device__ __forceinline__ float hmax16(float v) {
  v = fmaxf(v, __shfl_xor(v, 1)); v = fmaxf(v, __shfl_xor(v, 2));
  v = fmaxf(v, __shfl_xor(v, 4)); v = fmaxf(v, __shfl_xor(v, 8));
  return v;
}
__device__ __forceinline__ float hsum16(float v) {
  v += __shfl_xor(v, 1); v += __shfl_xor(v, 2);
  v += __shfl_xor(v, 4); v += __shfl_xor(v, 8);
  return v;
}
__device__ __forceinline__ float wsum32(float v) { v += __shfl_xor(v, 16); return hsum16(v); }

__device__ __forceinline__ float gelu_exact(float v) {
  return 0.5f * v * (1.0f + erff(v * 0.70710678118654752f));
}

// ---------------- weight cast: fp32 [K][N] -> f16 transposed [N][K] ----------------
__global__ __launch_bounds__(256) void k_cast_t(const float* __restrict__ src,
                                                f16* __restrict__ dst, int K, int N) {
  int tid = blockIdx.x * blockDim.x + threadIdx.x;
  if (tid >= K * N) return;
  int n = tid / K, k = tid - n * K;
  dst[tid] = (f16)src[(size_t)k * N + n];
}

// ---------------- LayerNorm (fp32 in) -> f16 out, one wave per row ----------------
__global__ __launch_bounds__(256) void k_layernorm(const float* __restrict__ x,
                                                   const float* __restrict__ g,
                                                   const float* __restrict__ bta,
                                                   f16* __restrict__ out, int M) {
  int row  = blockIdx.x * 8 + (threadIdx.x >> 5);
  int lane = threadIdx.x & 31;
  if (row >= M) return;
  const float* xr = x + (size_t)row * HIDN;
  float v[8], s = 0.f;
#pragma unroll
  for (int i = 0; i < 8; ++i) { v[i] = xr[lane * 8 + i]; s += v[i]; }
  float mean = wsum32(s) * (1.0f / HIDN);
  float var = 0.f;
#pragma unroll
  for (int i = 0; i < 8; ++i) { float d = v[i] - mean; var += d * d; }
  var = wsum32(var) * (1.0f / HIDN);
  float rstd = rsqrtf(var + 1e-5f);
#pragma unroll
  for (int i = 0; i < 8; ++i) {
    int c = lane * 8 + i;
    out[(size_t)row * HIDN + c] = (f16)((v[i] - mean) * rstd * g[c] + bta[c]);
  }
}

// ---------------- fused QKV GEMM: nx[4096x256] * WqkvT[768x256]^T ----------------
// epilogue scatters Q (pre-scaled by 1/sqrt(D)) and K as [bh][s][d], V as [bh][d][s]
__global__ __launch_bounds__(256) void k_gemm_qkv(const f16* __restrict__ A,
                                                  const f16* __restrict__ Wt,
                                                  const float* __restrict__ bq,
                                                  const float* __restrict__ bk,
                                                  const float* __restrict__ bv,
                                                  f16* __restrict__ Qb,
                                                  f16* __restrict__ Kb,
                                                  f16* __restrict__ Vt) {
  const int Nt = (3 * HIDN) / 32;                  // 24 column tiles
  int wave = threadIdx.x >> 5, lane = threadIdx.x & 31;
  int tile = blockIdx.x * 8 + wave;
  if (tile >= (MROWS / 16) * Nt) return;
  int tm = tile / Nt, tn = tile - tm * Nt;
  const f16* Ab = A + (size_t)tm * 16 * HIDN;
  v8f c0 = {}, c1 = {};
#pragma unroll
  for (int k = 0; k < HIDN; k += 32) {
    v16h a  = load_a_frag(Ab, HIDN, k, lane);
    v16h b0 = load_b_frag(Wt, HIDN, tn * 32, k, lane);
    v16h b1 = load_b_frag(Wt, HIDN, tn * 32 + 16, k, lane);
    c0 = wmma16(a, b0, c0);
    c1 = wmma16(a, b1, c1);
  }
  int n0 = tn * 32 + (lane & 15), n1 = n0 + 16;
  int hi = lane >> 4;
#pragma unroll
  for (int i = 0; i < 8; ++i) {
    int m = tm * 16 + i + (hi << 3);
    int b = m >> 11, s = m & (SEQ - 1);
#pragma unroll
    for (int half = 0; half < 2; ++half) {
      int   n   = half ? n1 : n0;
      float val = half ? c1[i] : c0[i];
      int grp = n >> 8, c = n & 255;
      int h = c >> 5, d = c & 31;
      size_t bh = (size_t)b * HEADS + h;
      if (grp == 0)
        Qb[(bh * SEQ + s) * HEAD_DIM + d] = (f16)((val + bq[c]) * 0.17677669529663689f);
      else if (grp == 1)
        Kb[(bh * SEQ + s) * HEAD_DIM + d] = (f16)(val + bk[c]);
      else
        Vt[(bh * HEAD_DIM + d) * SEQ + s] = (f16)(val + bv[c]);
    }
  }
}

// ---------------- masked flash attention: one wave per 16-query tile ----------------
__global__ __launch_bounds__(256) void k_attn(const f16* __restrict__ Qb,
                                              const f16* __restrict__ Kb,
                                              const f16* __restrict__ Vt,
                                              const unsigned char* __restrict__ mask,
                                              f16* __restrict__ attnOut) {
  __shared__ f16 Pl[8][16 * 32];                   // per-wave P staging tile
  int wave = threadIdx.x >> 5, lane = threadIdx.x & 31;
  int tile = blockIdx.x * 8 + wave;                // 2048 tiles total
  if (tile >= BATCH * HEADS * (SEQ / 16)) return;
  int qt = tile & (SEQ / 16 - 1);
  int bh = tile >> 7;
  int b = bh >> 3, h = bh & 7;

  const f16* Q  = Qb + (size_t)bh * SEQ * HEAD_DIM;
  const f16* Kp = Kb + (size_t)bh * SEQ * HEAD_DIM;
  const f16* Vp = Vt + (size_t)bh * HEAD_DIM * SEQ;
  const unsigned char* mk = mask + (size_t)b * SEQ * SEQ;

  int hi = lane >> 4, col = lane & 15;
  v16h aQ = load_a_frag(Q + (size_t)qt * 16 * HEAD_DIM, HEAD_DIM, 0, lane);

  float mi[8], li[8];
  v8f o0 = {}, o1 = {};
#pragma unroll
  for (int i = 0; i < 8; ++i) { mi[i] = NEGINF; li[i] = 0.f; }

  for (int kc = 0; kc < SEQ; kc += 32) {
    if (kc + 32 < SEQ)  // pull next K chunk toward L2 (global_prefetch_b8)
      __builtin_prefetch(Kp + (size_t)(kc + 32 + col) * HEAD_DIM, 0, 1);

    // scores = Q * K^T for 32 keys (K-dim = HEAD_DIM = 32 in one WMMA)
    v16h bk0 = load_b_frag(Kp, HEAD_DIM, kc,      0, lane);
    v16h bk1 = load_b_frag(Kp, HEAD_DIM, kc + 16, 0, lane);
    v8f z = {};
    v8f s0 = wmma16(aQ, bk0, z);
    v8f s1 = wmma16(aQ, bk1, z);

    // adjacency mask -> -1e9 (matches reference where-fill)
#pragma unroll
    for (int i = 0; i < 8; ++i) {
      int q = qt * 16 + i + (hi << 3);
      const unsigned char* mrow = mk + (size_t)q * SEQ + kc;
      if (!mrow[col])      s0[i] = NEGINF;
      if (!mrow[col + 16]) s1[i] = NEGINF;
    }

    // online softmax; rows stripe across 16-lane halves
    _Float16* P = &Pl[wave][0];
#pragma unroll
    for (int i = 0; i < 8; ++i) {
      float mx = hmax16(fmaxf(s0[i], s1[i]));
      float mn = fmaxf(mi[i], mx);
      float sc = __expf(mi[i] - mn);
      float p0 = __expf(s0[i] - mn);
      float p1 = __expf(s1[i] - mn);
      float rs = hsum16(p0 + p1);
      li[i] = li[i] * sc + rs;
      mi[i] = mn;
      o0[i] *= sc; o1[i] *= sc;
      int r = i + (hi << 3);
      P[r * 32 + col]      = (f16)p0;   // C-layout -> row-major 16x32 via LDS
      P[r * 32 + col + 16] = (f16)p1;
    }

    // O += P * V  (two 16x16 WMMAs over the 32-key chunk; V stored transposed)
    v16h aP  = load_a_frag(P, 32, 0, lane);
    v16h bv0 = load_b_frag(Vp, SEQ, 0,  kc, lane);
    v16h bv1 = load_b_frag(Vp, SEQ, 16, kc, lane);
    o0 = wmma16(aP, bv0, o0);
    o1 = wmma16(aP, bv1, o1);
  }

#pragma unroll
  for (int i = 0; i < 8; ++i) {
    int q = qt * 16 + i + (hi << 3);
    float inv = 1.0f / li[i];
    size_t base = ((size_t)b * SEQ + q) * HIDN + h * HEAD_DIM;
    attnOut[base + col]      = (f16)(o0[i] * inv);
    attnOut[base + col + 16] = (f16)(o1[i] * inv);
  }
}

// ---------------- generic WMMA GEMM: C = A[MxK]f16 * Wt[NxK]f16 + bias (+gelu)(+resid) ----------------
template <bool GELU>
__global__ __launch_bounds__(256) void k_gemm(const f16* __restrict__ A,
                                              const f16* __restrict__ Wt,
                                              const float* __restrict__ bias,
                                              const float* __restrict__ resid,
                                              float* __restrict__ outF,
                                              f16* __restrict__ outH,
                                              int M, int N, int K) {
  int wave = threadIdx.x >> 5, lane = threadIdx.x & 31;
  int Nt = N / 32;
  int tile = blockIdx.x * 8 + wave;
  if (tile >= (M / 16) * Nt) return;
  int tm = tile / Nt, tn = tile - tm * Nt;
  const f16* Ab = A + (size_t)tm * 16 * K;
  v8f c0 = {}, c1 = {};
  for (int k = 0; k < K; k += 32) {
    v16h a  = load_a_frag(Ab, K, k, lane);
    v16h b0 = load_b_frag(Wt, K, tn * 32, k, lane);
    v16h b1 = load_b_frag(Wt, K, tn * 32 + 16, k, lane);
    c0 = wmma16(a, b0, c0);
    c1 = wmma16(a, b1, c1);
  }
  int n0 = tn * 32 + (lane & 15), n1 = n0 + 16;
  int hi = lane >> 4;
#pragma unroll
  for (int i = 0; i < 8; ++i) {
    int m = tm * 16 + i + (hi << 3);
    float v0 = c0[i] + bias[n0];
    float v1 = c1[i] + bias[n1];
    if (GELU) { v0 = gelu_exact(v0); v1 = gelu_exact(v1); }
    if (resid) { v0 += resid[(size_t)m * N + n0]; v1 += resid[(size_t)m * N + n1]; }
    if (outF) { outF[(size_t)m * N + n0] = v0; outF[(size_t)m * N + n1] = v1; }
    if (outH) { outH[(size_t)m * N + n0] = (f16)v0; outH[(size_t)m * N + n1] = (f16)v1; }
  }
}

// ---------------- host orchestration ----------------
extern "C" void kernel_launch(void* const* d_in, const int* in_sizes, int n_in,
                              void* d_out, int out_size, void* d_ws, size_t ws_size,
                              hipStream_t stream) {
  const float* x     = (const float*)d_in[0];
  const unsigned char* mask = (const unsigned char*)d_in[1];  // bool adjacency
  const float* ln1_g = (const float*)d_in[2];
  const float* ln1_b = (const float*)d_in[3];
  const float* ln2_g = (const float*)d_in[4];
  const float* ln2_b = (const float*)d_in[5];
  const float* Wq = (const float*)d_in[6];   const float* bq = (const float*)d_in[7];
  const float* Wk = (const float*)d_in[8];   const float* bk = (const float*)d_in[9];
  const float* Wv = (const float*)d_in[10];  const float* bv = (const float*)d_in[11];
  const float* Wo = (const float*)d_in[12];  const float* bo = (const float*)d_in[13];
  const float* W1 = (const float*)d_in[14];  const float* b1 = (const float*)d_in[15];
  const float* W2 = (const float*)d_in[16];  const float* b2 = (const float*)d_in[17];
  float* out = (float*)d_out;

  // workspace carve-up (~25.5 MB total, fully rewritten every call)
  char* ws = (char*)d_ws;
  size_t off = 0;
  auto carve = [&](size_t bytes) { void* p = ws + off; off += (bytes + 255) & ~(size_t)255; return p; };
  f16* WqkvT  = (f16*)carve((size_t)3 * HIDN * HIDN * 2);   // [768][256]
  f16* WoT    = (f16*)carve((size_t)HIDN * HIDN * 2);       // [256][256]
  f16* W1T    = (f16*)carve((size_t)FF * HIDN * 2);         // [1024][256]
  f16* W2T    = (f16*)carve((size_t)HIDN * FF * 2);         // [256][1024]
  f16* nx1    = (f16*)carve((size_t)MROWS * HIDN * 2);
  f16* Qb     = (f16*)carve((size_t)BATCH * HEADS * SEQ * HEAD_DIM * 2);
  f16* Kb     = (f16*)carve((size_t)BATCH * HEADS * SEQ * HEAD_DIM * 2);
  f16* Vt     = (f16*)carve((size_t)BATCH * HEADS * HEAD_DIM * SEQ * 2);
  f16* attnO  = (f16*)carve((size_t)MROWS * HIDN * 2);
  float* x1   = (float*)carve((size_t)MROWS * HIDN * 4);
  f16* nx2    = (f16*)carve((size_t)MROWS * HIDN * 2);
  f16* hbuf   = (f16*)carve((size_t)MROWS * FF * 2);
  (void)ws_size; (void)n_in; (void)in_sizes; (void)out_size;

  // 1) cast + transpose all weights to f16 [N][K]
  k_cast_t<<<(HIDN * HIDN + 255) / 256, 256, 0, stream>>>(Wq, WqkvT,                 HIDN, HIDN);
  k_cast_t<<<(HIDN * HIDN + 255) / 256, 256, 0, stream>>>(Wk, WqkvT + HIDN * HIDN,   HIDN, HIDN);
  k_cast_t<<<(HIDN * HIDN + 255) / 256, 256, 0, stream>>>(Wv, WqkvT + 2 * HIDN * HIDN, HIDN, HIDN);
  k_cast_t<<<(HIDN * HIDN + 255) / 256, 256, 0, stream>>>(Wo, WoT,  HIDN, HIDN);
  k_cast_t<<<(HIDN * FF   + 255) / 256, 256, 0, stream>>>(W1, W1T,  HIDN, FF);
  k_cast_t<<<(FF   * HIDN + 255) / 256, 256, 0, stream>>>(W2, W2T,  FF,   HIDN);

  // 2) LN1 -> f16
  k_layernorm<<<MROWS / 8, 256, 0, stream>>>(x, ln1_g, ln1_b, nx1, MROWS);

  // 3) fused QKV projection (6144 wave-tiles)
  k_gemm_qkv<<<(MROWS / 16) * (3 * HIDN / 32) / 8, 256, 0, stream>>>(
      nx1, WqkvT, bq, bk, bv, Qb, Kb, Vt);

  // 4) masked flash attention (2048 wave-tiles)
  k_attn<<<BATCH * HEADS * (SEQ / 16) / 8, 256, 0, stream>>>(Qb, Kb, Vt, mask, attnO);

  // 5) O projection + bias + residual(x) -> x1 (fp32)
  k_gemm<false><<<(MROWS / 16) * (HIDN / 32) / 8, 256, 0, stream>>>(
      attnO, WoT, bo, x, x1, (f16*)nullptr, MROWS, HIDN, HIDN);

  // 6) LN2 -> f16
  k_layernorm<<<MROWS / 8, 256, 0, stream>>>(x1, ln2_g, ln2_b, nx2, MROWS);

  // 7) MLP up + exact GELU -> f16
  k_gemm<true><<<(MROWS / 16) * (FF / 32) / 8, 256, 0, stream>>>(
      nx2, W1T, b1, (const float*)nullptr, (float*)nullptr, hbuf, MROWS, FF, HIDN);

  // 8) MLP down + bias + residual(x1) -> d_out (fp32)
  k_gemm<false><<<(MROWS / 16) * (HIDN / 32) / 8, 256, 0, stream>>>(
      hbuf, W2T, b2, x1, out, (f16*)nullptr, MROWS, HIDN, FF);
}
